// SimpleModel2_44753559224988
// MI455X (gfx1250) — compile-verified, matrix operand order checked by reading, not run on hardware
//
#include <hip/hip_runtime.h>
#include <math.h>

// ---------------------------------------------------------------------------
// Types for CDNA5 WMMA (wave32)
// ---------------------------------------------------------------------------
typedef __bf16 bf16_t;
typedef __attribute__((ext_vector_type(16))) __bf16 v16bf;
typedef __attribute__((ext_vector_type(8)))  __bf16 bf16x8;
typedef __attribute__((ext_vector_type(8)))  float  v8f;

// Model dims
#define T_STEPS 256
#define BATCH   64
#define INDIM   512
#define HID     1024
#define G4      4096   // 4*HID
#define FC1OUT  512    // HID/2
#define FC2K    131072 // T_STEPS*FC1OUT

// ---------------------------------------------------------------------------
// Fragment loader: 16-bit A/B fragment = two contiguous 8-element (16B) chunks
// per lane: elems 0..7 at k0 + khalf, elems 8..15 at k0 + khalf + 16,
// where khalf = 8*(lane>=16).  rowbase points at row start + k0.
// ---------------------------------------------------------------------------
__device__ __forceinline__ v16bf load_frag(const bf16_t* __restrict__ rowbase,
                                           int khalf) {
  v16bf r;
  bf16x8* p = reinterpret_cast<bf16x8*>(&r);
  p[0] = *reinterpret_cast<const bf16x8*>(rowbase + khalf);
  p[1] = *reinterpret_cast<const bf16x8*>(rowbase + khalf + 16);
  return r;
}

__device__ __forceinline__ float sigmoidf_(float x) {
  return 1.0f / (1.0f + __expf(-x));
}

// ---------------------------------------------------------------------------
// Kernel 1: prep — bf16 conversions + transposes + state init.
// Flat index covers 8388608 threads (size of x).
// ---------------------------------------------------------------------------
__global__ void k_prep(const float* __restrict__ x,
                       const float* __restrict__ h0,
                       const float* __restrict__ c0,
                       const float* __restrict__ W,
                       const float* __restrict__ U,
                       const float* __restrict__ fc1_w,
                       bf16_t* __restrict__ xbf,
                       bf16_t* __restrict__ Wt,     // [4096][512]
                       bf16_t* __restrict__ Ut,     // [4096][1024]
                       bf16_t* __restrict__ fc1wt,  // [512][1024]
                       bf16_t* __restrict__ hbuf0,  // [64][1024]
                       float*  __restrict__ cstate) // [64][1024]
{
  const long i = (long)blockIdx.x * blockDim.x + threadIdx.x;

  // x -> bf16 (layout unchanged: [B][T][INDIM])
  if (i < (long)BATCH * T_STEPS * INDIM) xbf[i] = (bf16_t)x[i];

  // W (512 x 4096) -> Wt (4096 x 512), read-coalesced
  if (i < (long)INDIM * G4) {
    int k = (int)(i / G4), n = (int)(i % G4);
    Wt[(long)n * INDIM + k] = (bf16_t)W[i];
  }
  // U (1024 x 4096) -> Ut (4096 x 1024), read-coalesced
  if (i < (long)HID * G4) {
    int k = (int)(i / G4), n = (int)(i % G4);
    Ut[(long)n * HID + k] = (bf16_t)U[i];
  }
  // fc1_w (512 x 1024) already [n][k]; straight convert
  if (i < (long)FC1OUT * HID) fc1wt[i] = (bf16_t)fc1_w[i];

  // state init
  if (i < (long)BATCH * HID) {
    hbuf0[i]  = (bf16_t)h0[i];
    cstate[i] = c0[i];
  }
}

// ---------------------------------------------------------------------------
// Kernel 2: xw GEMM.  xw[m][n] = sum_k x[m][k]*W[k][n] + b[n]
// m = t*64 + b (row-major into xw [T][B][4096]); A row lives at
// xbf[(b*256 + t)*512].  One 16x16 tile per wave; K = 512 -> 16 WMMAs.
// Grid: 32768 blocks x 256 threads = 262144 waves = 1024 Mtiles * 256 Ntiles.
// ---------------------------------------------------------------------------
__global__ void k_xw_gemm(const bf16_t* __restrict__ xbf,
                          const bf16_t* __restrict__ Wt,
                          const float*  __restrict__ bias,
                          float* __restrict__ xw)
{
  const int lane   = threadIdx.x & 31;
  const int wave   = blockIdx.x * (blockDim.x >> 5) + (threadIdx.x >> 5);
  const int mtile  = wave >> 8;          // 0..1023
  const int ntile  = wave & 255;         // 0..255
  const int m0     = mtile << 4;
  const int n0     = ntile << 4;
  const int mlane  = lane & 15;
  const int khalf  = (lane >> 4) << 3;   // 0 or 8
  const int rowadd = (lane >> 4) << 3;   // C/D: M = r + 8*(lane>=16)

  const int mrow = m0 + mlane;           // A-fragment row
  const int ta = mrow >> 6, ba = mrow & 63;
  const bf16_t* arow = xbf + ((long)(ba * T_STEPS + ta)) * INDIM;
  const bf16_t* brow = Wt + (long)(n0 + mlane) * INDIM;

  v8f acc = {};
#pragma unroll
  for (int k0 = 0; k0 < INDIM; k0 += 32) {
    v16bf a  = load_frag(arow + k0, khalf);
    v16bf bb = load_frag(brow + k0, khalf);
    acc = __builtin_amdgcn_wmma_f32_16x16x32_bf16(false, a, false, bb,
                                                  (short)0, acc, false, false);
  }

  const int n  = n0 + mlane;
  const float bv = bias[n];
#pragma unroll
  for (int r = 0; r < 8; ++r) {
    const int m = m0 + r + rowadd;
    xw[(long)m * G4 + n] = acc[r] + bv;
  }
}

// ---------------------------------------------------------------------------
// Kernel 3: one LSTM time step (launched 256x).
// gates = xw[t] + h_in @ U ; each wave owns a 16(batch) x 16(hidden) tile and
// computes all 4 gate tiles so the nonlinearity is wave-local.
// Grid: 32 blocks x 256 threads = 256 waves = 4 Mtiles * 64 Jtiles.
// h ping-pong (bf16); c owned exclusively per wave (fp32).
// ---------------------------------------------------------------------------
__global__ void k_lstm_step(const float*  __restrict__ xw_t,  // [64][4096]
                            const bf16_t* __restrict__ hin,   // [64][1024]
                            bf16_t* __restrict__ hout,        // [64][1024]
                            float*  __restrict__ cstate,      // [64][1024]
                            const bf16_t* __restrict__ Ut,    // [4096][1024]
                            bf16_t* __restrict__ hs,          // [64][256][1024]
                            int t)
{
  const int lane   = threadIdx.x & 31;
  const int wave   = blockIdx.x * (blockDim.x >> 5) + (threadIdx.x >> 5);
  const int b0     = (wave >> 6) << 4;   // batch tile base (0,16,32,48)
  const int j0     = (wave & 63) << 4;   // hidden tile base
  const int mlane  = lane & 15;
  const int khalf  = (lane >> 4) << 3;
  const int rowadd = (lane >> 4) << 3;

  const bf16_t* arow  = hin + (long)(b0 + mlane) * HID;
  const bf16_t* br_i  = Ut + (long)(0 * HID + j0 + mlane) * HID;
  const bf16_t* br_f  = Ut + (long)(1 * HID + j0 + mlane) * HID;
  const bf16_t* br_g  = Ut + (long)(2 * HID + j0 + mlane) * HID;
  const bf16_t* br_o  = Ut + (long)(3 * HID + j0 + mlane) * HID;

  v8f ai = {}, af = {}, ag = {}, ao = {};
#pragma unroll 4
  for (int k0 = 0; k0 < HID; k0 += 32) {
    v16bf a = load_frag(arow + k0, khalf);
    v16bf bi = load_frag(br_i + k0, khalf);
    ai = __builtin_amdgcn_wmma_f32_16x16x32_bf16(false, a, false, bi, (short)0, ai, false, false);
    v16bf bf = load_frag(br_f + k0, khalf);
    af = __builtin_amdgcn_wmma_f32_16x16x32_bf16(false, a, false, bf, (short)0, af, false, false);
    v16bf bg = load_frag(br_g + k0, khalf);
    ag = __builtin_amdgcn_wmma_f32_16x16x32_bf16(false, a, false, bg, (short)0, ag, false, false);
    v16bf bo = load_frag(br_o + k0, khalf);
    ao = __builtin_amdgcn_wmma_f32_16x16x32_bf16(false, a, false, bo, (short)0, ao, false, false);
  }

  const int j = j0 + mlane;
#pragma unroll
  for (int r = 0; r < 8; ++r) {
    const int b = b0 + r + rowadd;
    const float* xwrow = xw_t + (long)b * G4;
    const float gi = ai[r] + xwrow[0 * HID + j];
    const float gf = af[r] + xwrow[1 * HID + j];
    const float gg = ag[r] + xwrow[2 * HID + j];
    const float go = ao[r] + xwrow[3 * HID + j];
    const float iv = sigmoidf_(gi);
    const float fv = sigmoidf_(gf);
    const float gv = tanhf(gg);
    const float ov = sigmoidf_(go);
    const long  ci = (long)b * HID + j;
    const float cn = fv * cstate[ci] + iv * gv;
    cstate[ci] = cn;
    const float hv = ov * tanhf(cn);
    hout[ci] = (bf16_t)hv;
    hs[((long)b * T_STEPS + t) * HID + j] = (bf16_t)hv;
  }
}

// ---------------------------------------------------------------------------
// Kernel 4: fc1 GEMM.  y[m][n] = sum_k hs[m][k]*fc1_w[n][k] + fc1_b[n]
// m = b*256 + t; y stored [m][512] == the fc2 flattening (b, t*512+n).
// Grid: 4096 blocks x 256 threads = 32768 waves = 1024 Mtiles * 32 Ntiles.
// ---------------------------------------------------------------------------
__global__ void k_fc1_gemm(const bf16_t* __restrict__ hs,
                           const bf16_t* __restrict__ fc1wt,
                           const float*  __restrict__ fc1b,
                           float* __restrict__ y)
{
  const int lane   = threadIdx.x & 31;
  const int wave   = blockIdx.x * (blockDim.x >> 5) + (threadIdx.x >> 5);
  const int mtile  = wave >> 5;          // 0..1023
  const int ntile  = wave & 31;          // 0..31
  const int m0     = mtile << 4;
  const int n0     = ntile << 4;
  const int mlane  = lane & 15;
  const int khalf  = (lane >> 4) << 3;
  const int rowadd = (lane >> 4) << 3;

  const bf16_t* arow = hs + (long)(m0 + mlane) * HID;
  const bf16_t* brow = fc1wt + (long)(n0 + mlane) * HID;

  v8f acc = {};
#pragma unroll 4
  for (int k0 = 0; k0 < HID; k0 += 32) {
    v16bf a  = load_frag(arow + k0, khalf);
    v16bf bb = load_frag(brow + k0, khalf);
    acc = __builtin_amdgcn_wmma_f32_16x16x32_bf16(false, a, false, bb,
                                                  (short)0, acc, false, false);
  }

  const int n = n0 + mlane;
  const float bv = fc1b[n];
#pragma unroll
  for (int r = 0; r < 8; ++r) {
    const int m = m0 + r + rowadd;
    y[(long)m * FC1OUT + n] = acc[r] + bv;
  }
}

// ---------------------------------------------------------------------------
// Kernel 5: fc2 — out[b][j] = dot(y[b, :131072], fc2_w[j]) + fc2_b[j]
// One block per batch row; LDS tree reduction.
// ---------------------------------------------------------------------------
__global__ void k_fc2(const float* __restrict__ y,
                      const float* __restrict__ fc2w,
                      const float* __restrict__ fc2b,
                      float* __restrict__ out)
{
  __shared__ float s0[256];
  __shared__ float s1[256];
  const int b   = blockIdx.x;
  const int tid = threadIdx.x;
  const float* yr = y + (long)b * FC2K;
  const float* w0 = fc2w;
  const float* w1 = fc2w + FC2K;

  float a0 = 0.f, a1 = 0.f;
  for (int i = tid; i < FC2K; i += 256) {
    const float v = yr[i];
    a0 += v * w0[i];
    a1 += v * w1[i];
  }
  s0[tid] = a0; s1[tid] = a1;
  __syncthreads();
  for (int s = 128; s > 0; s >>= 1) {
    if (tid < s) { s0[tid] += s0[tid + s]; s1[tid] += s1[tid + s]; }
    __syncthreads();
  }
  if (tid == 0) {
    out[b * 2 + 0] = s0[0] + fc2b[0];
    out[b * 2 + 1] = s1[0] + fc2b[1];
  }
}

// ---------------------------------------------------------------------------
// Launch
// ---------------------------------------------------------------------------
extern "C" void kernel_launch(void* const* d_in, const int* in_sizes, int n_in,
                              void* d_out, int out_size, void* d_ws, size_t ws_size,
                              hipStream_t stream) {
  const float* x     = (const float*)d_in[0];
  const float* h0    = (const float*)d_in[1];
  const float* c0    = (const float*)d_in[2];
  const float* W     = (const float*)d_in[3];
  const float* U     = (const float*)d_in[4];
  const float* bvec  = (const float*)d_in[5];
  const float* fc1_w = (const float*)d_in[6];
  const float* fc1_b = (const float*)d_in[7];
  const float* fc2_w = (const float*)d_in[8];
  const float* fc2_b = (const float*)d_in[9];
  float* out = (float*)d_out;

  // Workspace carve-up (256B aligned)
  char* ws = (char*)d_ws;
  size_t off = 0;
  auto alloc = [&](size_t bytes) {
    void* p = ws + off;
    off += (bytes + 255) & ~(size_t)255;
    return p;
  };
  float*  xw     = (float*) alloc((size_t)T_STEPS * BATCH * G4 * 4);   // 256 MB
  bf16_t* xbf    = (bf16_t*)alloc((size_t)BATCH * T_STEPS * INDIM * 2);
  bf16_t* Wt     = (bf16_t*)alloc((size_t)G4 * INDIM * 2);
  bf16_t* Ut     = (bf16_t*)alloc((size_t)G4 * HID * 2);
  bf16_t* fc1wt  = (bf16_t*)alloc((size_t)FC1OUT * HID * 2);
  bf16_t* hbuf0  = (bf16_t*)alloc((size_t)BATCH * HID * 2);
  bf16_t* hbuf1  = (bf16_t*)alloc((size_t)BATCH * HID * 2);
  float*  cstate = (float*) alloc((size_t)BATCH * HID * 4);
  bf16_t* hs     = (bf16_t*)alloc((size_t)BATCH * T_STEPS * HID * 2);
  float*  y      = (float*) alloc((size_t)BATCH * T_STEPS * FC1OUT * 4);

  // 1) conversions / transposes / state init
  k_prep<<<32768, 256, 0, stream>>>(x, h0, c0, W, U, fc1_w,
                                    xbf, Wt, Ut, fc1wt, hbuf0, cstate);

  // 2) xw = x @ W + b   (WMMA bf16)
  k_xw_gemm<<<32768, 256, 0, stream>>>(xbf, Wt, bvec, xw);

  // 3) sequential LSTM recurrence, ping-pong h
  for (int t = 0; t < T_STEPS; ++t) {
    const bf16_t* hin  = (t & 1) ? hbuf1 : hbuf0;
    bf16_t*       hout = (t & 1) ? hbuf0 : hbuf1;
    k_lstm_step<<<32, 256, 0, stream>>>(xw + (size_t)t * BATCH * G4,
                                        hin, hout, cstate, Ut, hs, t);
  }

  // 4) fc1 (WMMA bf16)
  k_fc1_gemm<<<4096, 256, 0, stream>>>(hs, fc1wt, fc1_b, y);

  // 5) fc2 reduction
  k_fc2<<<BATCH, 256, 0, stream>>>(y, fc2_w, fc2_b, out);
}
